// Attention_v2_3410204033251
// MI455X (gfx1250) — matrix operand order repaired and checked by reading, not compiled, not run on hardware
//
#include <hip/hip_runtime.h>

// ---------------------------------------------------------------------------
// CDNA5 (gfx1250) flash-attention pipeline, wave32 + v_wmma_f32_16x16x32_bf16.
// Round 2: LDS-free GEMMs (pre-transposed bf16 weights), LDS-free K/V path in
// attention (V produced transposed), DPP all-reduce instead of ds_bpermute,
// log2-domain softmax (exp2, scale folded into Q projection), global prefetch.
// ---------------------------------------------------------------------------

typedef __bf16 bf16;
typedef __attribute__((ext_vector_type(16))) __bf16 v16bf;
typedef __attribute__((ext_vector_type(8)))  float  v8f;

#define TQd     2048
#define TKd     16384
#define DMODEL  256
#define NHEAD   8
#define HDIM    32
#define KEYTILE 128

union AFrag { v16bf v; uint4 q[2]; };

__device__ __forceinline__ v8f wmma_bf16(v16bf a, v16bf b, v8f c) {
  return __builtin_amdgcn_wmma_f32_16x16x32_bf16(false, a, false, b,
                                                 (short)0, c, false, false);
}

// B fragment (32x16): per-lane 8 bf16 at p (K=kb..kb+7) and 8 at p+16 (K=kb+16..kb+23).
__device__ __forceinline__ v16bf load_b16x16(const bf16* p) {
  AFrag f;
  f.q[0] = *(const uint4*)(p);
  f.q[1] = *(const uint4*)(p + 16);
  return f.v;
}
// A fragment (16x32): per-lane 8 bf16 at p (K=base..base+7) and 8 at p+16
// (K=base+16..base+23), per ISA 7.12.2 16-bit A layout.
__device__ __forceinline__ v16bf load_a8_skip8(const bf16* p) {
  AFrag f;
  f.q[0] = *(const uint4*)(p);
  f.q[1] = *(const uint4*)(p + 16);
  return f.v;
}

// ---- DPP all-reduce across each 16-lane half of the wave32 ---------------
// quad_perm(xor1)=0xB1, quad_perm(xor2)=0x4E, row_ror:4=0x124, row_ror:8=0x128
template <int CTRL>
__device__ __forceinline__ float dpp_mv(float x) {
  return __builtin_bit_cast(
      float, __builtin_amdgcn_update_dpp(0, __builtin_bit_cast(int, x), CTRL,
                                         0xF, 0xF, true));
}
__device__ __forceinline__ float halfmax16(float v) {
  v = fmaxf(v, dpp_mv<0xB1>(v));
  v = fmaxf(v, dpp_mv<0x4E>(v));
  v = fmaxf(v, dpp_mv<0x124>(v));
  v = fmaxf(v, dpp_mv<0x128>(v));
  return v;
}
__device__ __forceinline__ float halfsum16(float v) {
  v += dpp_mv<0xB1>(v);
  v += dpp_mv<0x4E>(v);
  v += dpp_mv<0x124>(v);
  v += dpp_mv<0x128>(v);
  return v;
}

// ---------------------------------------------------------------------------
// One-shot prep: WT[n*256+k] = bf16(W[k*256+n])  (weights are 256 KB; cost ~0)
// ---------------------------------------------------------------------------
__global__ __launch_bounds__(256)
void transpose_w_bf16(const float* __restrict__ W, bf16* __restrict__ WT) {
  const int n = blockIdx.x, k = threadIdx.x;
  WT[n * DMODEL + k] = (bf16)W[k * DMODEL + n];
}

// ---------------------------------------------------------------------------
// LDS-free GEMM: out[M x 256] = (A[M x 256] @ W + bias) * scale
// WT is bf16 [N=256][K=256]. Block 128 = 4 waves; wave tile 16x64; no barriers.
// ---------------------------------------------------------------------------
template <bool A_BF16, bool OUT_F32, bool OUT_TRANS>
__global__ __launch_bounds__(128)
void gemm_bias_wmma(const void* __restrict__ A_, const bf16* __restrict__ WT,
                    const float* __restrict__ bias, void* __restrict__ out_,
                    int ldt, float scale) {
  const int tid = threadIdx.x, lane = tid & 31, wave = tid >> 5;
  const int halfsel = lane >> 4, ln16 = lane & 15;
  const int n0 = blockIdx.x * 64, m0 = blockIdx.y * 64;
  const int row = m0 + wave * 16 + ln16;
  const int abase = halfsel * 8, kb = halfsel * 8;

  v8f acc[4] = {};
#pragma unroll
  for (int kt = 0; kt < DMODEL; kt += 32) {
    v16bf a;
    if (A_BF16) {
      a = load_a8_skip8((const bf16*)A_ + row * DMODEL + kt + abase);
    } else {
      const float* ap = (const float*)A_ + row * DMODEL + kt + abase;
      AFrag f;
#pragma unroll
      for (int j = 0; j < 8; ++j) f.v[j] = (bf16)ap[j];
#pragma unroll
      for (int j = 0; j < 8; ++j) f.v[8 + j] = (bf16)ap[16 + j];
      a = f.v;
    }
#pragma unroll
    for (int nt = 0; nt < 4; ++nt) {
      const int n = n0 + nt * 16 + ln16;
      v16bf b = load_b16x16(WT + n * DMODEL + kt + kb);
      acc[nt] = wmma_bf16(a, b, acc[nt]);
    }
  }
#pragma unroll
  for (int nt = 0; nt < 4; ++nt) {
#pragma unroll
    for (int r = 0; r < 8; ++r) {
      const int rr = m0 + wave * 16 + r + 8 * halfsel;
      const int cc = n0 + nt * 16 + ln16;
      const float v = (acc[nt][r] + bias[cc]) * scale;
      if (OUT_TRANS)    ((bf16*)out_)[cc * ldt + rr] = (bf16)v;   // V^T path
      else if (OUT_F32) ((float*)out_)[rr * DMODEL + cc] = v;
      else              ((bf16*)out_)[rr * DMODEL + cc] = (bf16)v;
    }
  }
}

// ---------------------------------------------------------------------------
// Flash attention: grid (TQ/64, H), block 128 (4 waves x 16 q-rows).
// Key tile 128. K fragments load straight from row-major kh; V fragments load
// straight from transposed vhT[H*VD][T] -- both contiguous 2x b128 per lane,
// L2-resident. Only LDS use: wave-private P C-layout -> A-layout round-trip.
// Softmax in log2 domain; log2(e)/sqrt(32) pre-folded into Q.
// ---------------------------------------------------------------------------
__global__ __launch_bounds__(128)
void flash_attn_wmma(const bf16* __restrict__ qh, const bf16* __restrict__ kh,
                     const bf16* __restrict__ vhT, bf16* __restrict__ attn) {
  __shared__ __align__(16) bf16 Ps[4][16][144];  // row stride 288B (16B mult.)

  const int tid = threadIdx.x, lane = tid & 31, wave = tid >> 5;
  const int halfsel = lane >> 4, ln16 = lane & 15;
  const int h = blockIdx.y, q0 = blockIdx.x * 64;
  const int abase = halfsel * 8, kb = halfsel * 8;

  const v16bf qa =
      load_a8_skip8(qh + (q0 + wave * 16 + ln16) * DMODEL + h * HDIM + abase);

  float runMax[8], runSum[8];
#pragma unroll
  for (int r = 0; r < 8; ++r) { runMax[r] = -3.0e38f; runSum[r] = 0.0f; }
  v8f o0 = {}, o1 = {};

  for (int kt = 0; kt < TKd; kt += KEYTILE) {
    if (kt + KEYTILE < TKd) {  // warm next tile (global_prefetch_b8)
      __builtin_prefetch(kh + (kt + KEYTILE + lane) * DMODEL + h * HDIM, 0, 3);
      __builtin_prefetch(vhT + (h * HDIM + lane) * TKd + kt + KEYTILE, 0, 3);
    }

    // --- S = Q @ K^T : 8 x 16x16 score tiles, one WMMA each ---------------
    v8f s[8];
#pragma unroll
    for (int nt = 0; nt < 8; ++nt) {
      v16bf b = load_b16x16(kh + (kt + nt * 16 + ln16) * DMODEL + h * HDIM + kb);
      v8f z = {};
      s[nt] = wmma_bf16(qa, b, z);
    }

    // --- online softmax (log2 domain); rows r+8*halfsel across ln16 lanes --
#pragma unroll
    for (int r = 0; r < 8; ++r) {
      float tmax = s[0][r];
#pragma unroll
      for (int nt = 1; nt < 8; ++nt) tmax = fmaxf(tmax, s[nt][r]);
      tmax = halfmax16(tmax);
      const float nm = fmaxf(runMax[r], tmax);
      const float corr = exp2f(runMax[r] - nm);
      runMax[r] = nm;
      float psum = 0.0f;
#pragma unroll
      for (int nt = 0; nt < 8; ++nt) {
        const float p = exp2f(s[nt][r] - nm);
        s[nt][r] = p;
        psum += p;
      }
      psum = halfsum16(psum);
      runSum[r] = runSum[r] * corr + psum;
      o0[r] *= corr;
      o1[r] *= corr;
    }

    // --- P: C-layout -> LDS -> A-layout (wave-private buffer) --------------
#pragma unroll
    for (int nt = 0; nt < 8; ++nt)
#pragma unroll
      for (int r = 0; r < 8; ++r)
        Ps[wave][r + 8 * halfsel][nt * 16 + ln16] = (bf16)s[nt][r];
    asm volatile("s_wait_dscnt 0" ::: "memory");  // in-wave cross-lane RAW

    // --- O += P(16x128) @ V(128x32) : 4 K-chunks x 2 column tiles ----------
#pragma unroll
    for (int kc = 0; kc < 4; ++kc) {
      v16bf pa = load_a8_skip8(&Ps[wave][ln16][kc * 32 + abase]);
      v16bf vb0 =
          load_b16x16(vhT + (h * HDIM + ln16) * TKd + kt + kc * 32 + kb);
      v16bf vb1 =
          load_b16x16(vhT + (h * HDIM + 16 + ln16) * TKd + kt + kc * 32 + kb);
      o0 = wmma_bf16(pa, vb0, o0);
      o1 = wmma_bf16(pa, vb1, o1);
    }
  }

  // --- normalize, emit bf16 attn [TQ, H*VD] for the output projection ------
#pragma unroll
  for (int r = 0; r < 8; ++r) {
    const float inv = 1.0f / runSum[r];
    const int row = q0 + wave * 16 + r + 8 * halfsel;
    attn[row * DMODEL + h * HDIM + ln16]      = (bf16)(o0[r] * inv);
    attn[row * DMODEL + h * HDIM + 16 + ln16] = (bf16)(o1[r] * inv);
  }
}

// ---------------------------------------------------------------------------
extern "C" void kernel_launch(void* const* d_in, const int* in_sizes, int n_in,
                              void* d_out, int out_size, void* d_ws,
                              size_t ws_size, hipStream_t stream) {
  (void)in_sizes; (void)n_in; (void)out_size; (void)ws_size;
  const float* query = (const float*)d_in[0];
  const float* key_  = (const float*)d_in[1];
  const float* value = (const float*)d_in[2];
  const float* wq = (const float*)d_in[3];
  const float* bq = (const float*)d_in[4];
  const float* wk = (const float*)d_in[5];
  const float* bk = (const float*)d_in[6];
  const float* wv = (const float*)d_in[7];
  const float* bv = (const float*)d_in[8];
  const float* wo = (const float*)d_in[9];
  const float* bo = (const float*)d_in[10];
  float* out = (float*)d_out;

  // workspace carve-out (all bf16; ~19 MB total, L2-resident)
  char* ws = (char*)d_ws;
  bf16* qh  = (bf16*)ws; ws += (size_t)TQd * DMODEL * sizeof(bf16);
  bf16* kh  = (bf16*)ws; ws += (size_t)TKd * DMODEL * sizeof(bf16);
  bf16* vhT = (bf16*)ws; ws += (size_t)TKd * DMODEL * sizeof(bf16);  // [H*VD][T]
  bf16* at  = (bf16*)ws; ws += (size_t)TQd * DMODEL * sizeof(bf16);
  bf16* wqT = (bf16*)ws; ws += (size_t)DMODEL * DMODEL * sizeof(bf16);
  bf16* wkT = (bf16*)ws; ws += (size_t)DMODEL * DMODEL * sizeof(bf16);
  bf16* wvT = (bf16*)ws; ws += (size_t)DMODEL * DMODEL * sizeof(bf16);
  bf16* woT = (bf16*)ws;

  // bf16-transposed weights (one-shot, 256 KB each)
  transpose_w_bf16<<<DMODEL, DMODEL, 0, stream>>>(wq, wqT);
  transpose_w_bf16<<<DMODEL, DMODEL, 0, stream>>>(wk, wkT);
  transpose_w_bf16<<<DMODEL, DMODEL, 0, stream>>>(wv, wvT);
  transpose_w_bf16<<<DMODEL, DMODEL, 0, stream>>>(wo, woT);

  // log2(e)/sqrt(32): puts softmax in the exp2 domain with no per-score mul
  const float qscale = 0.25503482459285448f;

  gemm_bias_wmma<false, false, false><<<dim3(4, TQd / 64), 128, 0, stream>>>(
      query, wqT, bq, qh, 0, qscale);
  gemm_bias_wmma<false, false, false><<<dim3(4, TKd / 64), 128, 0, stream>>>(
      key_, wkT, bk, kh, 0, 1.0f);
  gemm_bias_wmma<false, false, true><<<dim3(4, TKd / 64), 128, 0, stream>>>(
      value, wvT, bv, vhT, TKd, 1.0f);

  flash_attn_wmma<<<dim3(TQd / 64, NHEAD), 128, 0, stream>>>(qh, kh, vhT, at);

  gemm_bias_wmma<true, true, false><<<dim3(4, TQd / 64), 128, 0, stream>>>(
      at, woT, bo, out, 0, 1.0f);
}
